// Core_28527172780445
// MI455X (gfx1250) — matrix operand order repaired and checked by reading, not compiled, not run on hardware
//
#include <hip/hip_runtime.h>

// ---------------------------------------------------------------------------
// 2-layer LSTM (B=32, T=1068, H=256) + dense(273408->256) for gfx1250.
//   1) pack W1,U1,W2,U2 fp32 -> bf16 in WMMA-B fragment-major layout
//      (lane-contiguous 32B per fragment -> b128 loads, no shuffle movs)
//   2) XW1 = X@W1 + b1  (WMMA GEMM, C stored fragment-major as v8f)
//   3) recurrence L1    (persistent 16 blocks, U slice in LDS pre-swizzled,
//                        h double-buffered in global, grid barrier per step)
//   4) XW2 = seq1@W2 + b2
//   5) recurrence L2    (init state = (h1,h1) per reference quirk)
//   6) dense: partial-K WMMA, one Wd pass shared by both M-tiles, atomics
// All matmuls: v_wmma_f32_16x16x32_bf16 (fp32 accumulate).
// ---------------------------------------------------------------------------

typedef __attribute__((ext_vector_type(16))) __bf16 v16bf;
typedef __attribute__((ext_vector_type(2)))  __bf16 bf16x2;
typedef __attribute__((ext_vector_type(8)))  float  v8f;

#define T_STEPS 1068
#define Hdim    256
#define FH      1024               // 4*H
#define Bdim    32
#define BH      8192               // B*H
#define MROWS   (Bdim * T_STEPS)   // 34176
#define MTILES  (MROWS / 16)       // 2136
#define KDENSE  (T_STEPS * Hdim)   // 273408
#define RBLK    16                 // persistent blocks in recurrence

// ---------------- WMMA fragment helpers (contiguous k = 16*(lane/16) + e, A and B alike) ----------------

// A 16x32 bf16 (MxK), row-major source, row = lane&15.
// lanes 0-15: r0..7 -> K=0..15   (contiguous 16-elem run)
// lanes16-31: r0..7 -> K=16..31
__device__ inline v16bf load_a_frag(const __bf16* __restrict__ base,
                                    size_t rowStride, int lane) {
  const __bf16* rp = base + (size_t)(lane & 15) * rowStride;
  const int kb = (lane & 16) ? 16 : 0;
  v16bf a;
#pragma unroll
  for (int r = 0; r < 8; ++r) {
    int k = kb + 2 * r;
    bf16x2 p = *reinterpret_cast<const bf16x2*>(rp + k);   // merges into b128
    a[2 * r]     = p.x;
    a[2 * r + 1] = p.y;
  }
  return a;
}

// Packed (fragment-major) B operand: one 32B vector per lane.
__device__ inline v16bf load_b_packed(const __bf16* __restrict__ frag, int lane) {
  return *reinterpret_cast<const v16bf*>(frag + (size_t)lane * 16);
}

// B 32x16 bf16 (KxN) from row-major fp32 [K][N], converted on the fly.
// lanes 0-15 hold K=0..15 (2 per VGPR), lanes 16-31 hold K=16..31.
__device__ inline v16bf load_b_frag_f32(const float* __restrict__ base,
                                        size_t rowStride, int lane) {
  const float* cp = base + (lane & 15);
  const int kb = (lane & 16) ? 16 : 0;
  v16bf b;
#pragma unroll
  for (int r = 0; r < 8; ++r) {
    int k = kb + 2 * r;
    b[2 * r]     = (__bf16)cp[(size_t)k * rowStride];
    b[2 * r + 1] = (__bf16)cp[(size_t)(k + 1) * rowStride];
  }
  return b;
}

__device__ inline float sigmoidf_(float x) { return 1.0f / (1.0f + __expf(-x)); }

// ---------------- grid barrier (persistent recurrence kernel) ----------------

__device__ inline void grid_barrier(int* cnt, int* gen, int nblk) {
  __syncthreads();
  if (threadIdx.x == 0) {
    __threadfence();
    int g = __hip_atomic_load(gen, __ATOMIC_RELAXED, __HIP_MEMORY_SCOPE_AGENT);
    if (atomicAdd(cnt, 1) == nblk - 1) {
      __hip_atomic_store(cnt, 0, __ATOMIC_RELAXED, __HIP_MEMORY_SCOPE_AGENT);
      __hip_atomic_store(gen, g + 1, __ATOMIC_RELEASE, __HIP_MEMORY_SCOPE_AGENT);
    } else {
      while (__hip_atomic_load(gen, __ATOMIC_ACQUIRE, __HIP_MEMORY_SCOPE_AGENT) == g)
        __builtin_amdgcn_s_sleep(2);
    }
    __threadfence();
  }
  __syncthreads();
}

// ------------------------------- kernels ------------------------------------

__global__ void cvt_f32_bf16(const float* __restrict__ in,
                             __bf16* __restrict__ out, int n) {
  for (int i = blockIdx.x * blockDim.x + threadIdx.x; i < n;
       i += gridDim.x * blockDim.x)
    out[i] = (__bf16)in[i];
}

// Pack fp32 weight [256][1024] into WMMA-B fragment-major bf16:
//   Wp[ct(64)][kt(8)][lane(32)][e(16)],  k = kt*32 + (lane&16?16:0) + e,
//                                        n = ct*16 + (lane&15)
__global__ void pack_w(const float* __restrict__ W, __bf16* __restrict__ Wp) {
  int idx = blockIdx.x * 256 + threadIdx.x;          // 262144 total
  if (idx >= 64 * 8 * 32 * 16) return;
  int e    = idx & 15;
  int lane = (idx >> 4) & 31;
  int kt   = (idx >> 9) & 7;
  int ct   = idx >> 12;
  int k = kt * 32 + ((lane & 16) ? 16 : 0) + e;
  int n = ct * 16 + (lane & 15);
  Wp[idx] = (__bf16)W[k * FH + n];
}

__global__ void zero_init(float* __restrict__ acc, int* __restrict__ bars) {
  int i = blockIdx.x * 256 + threadIdx.x;
  if (i < BH) acc[i] = 0.0f;
  if (i < 8)  bars[i] = 0;
}

__global__ void init_h(const float* __restrict__ h, __bf16* __restrict__ hbuf) {
  int i = blockIdx.x * 256 + threadIdx.x;
  if (i < BH) hbuf[i] = (__bf16)h[i];
}

// OUT fragment-major: XWp[mtg(2136)][ct(64)][lane(32)][8 fp32]  (v8f per lane)
// xlayout=1: A rows in X layout ( b*(T*H) + t*H ); xlayout=0: A row r at r*H.
__global__ void __launch_bounds__(256) gemm_xw(const __bf16* __restrict__ A,
                                               const __bf16* __restrict__ Wp,
                                               const float* __restrict__ bias,
                                               float* __restrict__ XWp,
                                               int xlayout) {
  const int wave = blockIdx.x * 8 + (threadIdx.x >> 5);
  const int mt = wave >> 6;        // 0..2135
  const int nt = wave & 63;        // col tile
  if (mt >= MTILES) return;
  const int lane = threadIdx.x & 31;

  v8f c;
  {
    float bv = bias[nt * 16 + (lane & 15)];
#pragma unroll
    for (int r = 0; r < 8; ++r) c[r] = bv;
  }

  const __bf16* abase;
  size_t astride;
  if (xlayout) {                   // 16-row tile: fixed t, batches b0..b0+15
    int t = mt >> 1, b0 = (mt & 1) * 16;
    abase   = A + (size_t)b0 * (T_STEPS * Hdim) + (size_t)t * Hdim;
    astride = (size_t)T_STEPS * Hdim;
  } else {                         // seq layout [T][B][H]
    abase   = A + (size_t)mt * 16 * Hdim;
    astride = Hdim;
  }

#pragma unroll
  for (int kt = 0; kt < 8; ++kt) {
    v16bf a = load_a_frag(abase + kt * 32, astride, lane);
    v16bf b = load_b_packed(Wp + (size_t)(nt * 8 + kt) * 512, lane);
    c = __builtin_amdgcn_wmma_f32_16x16x32_bf16(false, a, false, b,
                                                (short)0, c, false, false);
  }
  *reinterpret_cast<v8f*>(XWp + ((size_t)(mt * 64 + nt) * 32 + lane) * 8) = c;
}

// Persistent LSTM recurrence. 16 blocks x 256 threads; block `blk` owns hidden
// columns [blk*16, blk*16+16) => U col-tiles { g*16 + blk : g in 0..3 }.
__global__ void __launch_bounds__(256) lstm_rec(const float* __restrict__ XWp,  // packed frags
                                                const __bf16* __restrict__ Up,  // packed W-layout
                                                const float* __restrict__ c0,   // [32][256]
                                                __bf16* __restrict__ hbuf,      // [2][32][256]
                                                float* __restrict__ hT,
                                                float* __restrict__ cT,
                                                __bf16* __restrict__ seq,       // [T][32][256]
                                                int* cnt, int* gen) {
  __shared__ __bf16 Us[4 * 8 * 512];   // 4 col-tiles x 8 k-tiles, packed (32KB)
  __shared__ float  zs[32 * 64];
  __shared__ float  cs[32 * 16];

  const int blk  = blockIdx.x;
  const int tid  = threadIdx.x;
  const int wv   = tid >> 5;
  const int lane = tid & 31;

  // stage packed U col-tiles (straight vector copy: fragments stay packed)
  for (int v = tid; v < 1024; v += 256) {          // 1024 x v16bf = 32KB
    int g  = v >> 8;                               // gate
    int lv = v & 255;                              // vec within col-tile
    const v16bf* src = reinterpret_cast<const v16bf*>(
        Up + ((size_t)(g * 16 + blk) * 8) * 512 + (size_t)lv * 16);
    reinterpret_cast<v16bf*>(Us)[g * 256 + lv] = *src;
  }
  for (int i = tid; i < 32 * 16; i += 256) {
    int m = i >> 4, j = i & 15;
    cs[i] = c0[m * Hdim + blk * 16 + j];
  }
  __syncthreads();

  const int mt = wv >> 2;   // 0..1 : rows 0-15 / 16-31
  const int nt = wv & 3;    // 0..3 : gate index (i,f,g,o)

#pragma unroll 1
  for (int t = 0; t < T_STEPS; ++t) {
    const __bf16* hin = hbuf + (size_t)(t & 1) * BH;

    // C init = packed XW fragment (x@W + b, fp32): one v8f load
    v8f c = *reinterpret_cast<const v8f*>(
        XWp + ((size_t)((t * 2 + mt) * 64 + (nt * 16 + blk)) * 32 + lane) * 8);

#pragma unroll
    for (int kt = 0; kt < 8; ++kt) {
      v16bf a = load_a_frag(hin + (size_t)mt * 16 * Hdim + kt * 32, Hdim, lane);
      v16bf b = load_b_packed(Us + (size_t)(nt * 8 + kt) * 512, lane);
      c = __builtin_amdgcn_wmma_f32_16x16x32_bf16(false, a, false, b,
                                                  (short)0, c, false, false);
    }
    {
      int n = lane & 15, mb = (lane & 16) ? 8 : 0;
#pragma unroll
      for (int r = 0; r < 8; ++r)
        zs[(mt * 16 + mb + r) * 64 + nt * 16 + n] = c[r];
    }
    __syncthreads();

    // gates + state update for this block's 32x16 slice
    for (int i = tid; i < 512; i += 256) {
      int m = i >> 4, j = i & 15;
      float ig = sigmoidf_(zs[m * 64 +      j]);
      float fg = sigmoidf_(zs[m * 64 + 16 + j]);
      float gg = tanhf    (zs[m * 64 + 32 + j]);
      float og = sigmoidf_(zs[m * 64 + 48 + j]);
      float cn = fg * cs[i] + ig * gg;
      float hn = og * tanhf(cn);
      cs[i] = cn;
      int gcol = blk * 16 + j;
      __bf16 hb = (__bf16)hn;
      hbuf[(size_t)((t + 1) & 1) * BH + m * Hdim + gcol] = hb;
      seq[(size_t)t * BH + m * Hdim + gcol] = hb;
      if (t == T_STEPS - 1) {
        hT[m * Hdim + gcol] = hn;
        cT[m * Hdim + gcol] = cn;
      }
    }
    grid_barrier(cnt, gen, RBLK);   // publish h(t+1) to all blocks
  }
}

// Dense partial-K: grid (134 K-chunks, 16 n-tiles). Both M-tiles share one
// B fragment => Wd (280MB) streamed exactly once. fp32 atomic reduction.
__global__ void __launch_bounds__(256) dense_partial(const __bf16* __restrict__ A2, // [T][32][256]
                                                     const float* __restrict__ Wd,  // [K][256]
                                                     float* __restrict__ acc) {
  __shared__ float part[8 * 512];
  const int nt = blockIdx.y;                 // 0..15
  const int wv = threadIdx.x >> 5, lane = threadIdx.x & 31;

  v8f c0t = {}, c1t = {};
  const int kstart = blockIdx.x * 2048 + wv * 256;
#pragma unroll 1
  for (int s = 0; s < 8; ++s) {
    int kk = kstart + s * 32;
    if (kk < KDENSE) {   // wave-uniform guard: EXEC stays all-ones for WMMA
      // A[b][k] = seq2[t = k>>8][b][k&255]
      const __bf16* abase = A2 + (size_t)(kk >> 8) * BH + (kk & 255);
      v16bf a0 = load_a_frag(abase,              Hdim, lane);   // b =  0..15
      v16bf a1 = load_a_frag(abase + 16 * Hdim,  Hdim, lane);   // b = 16..31
      v16bf b  = load_b_frag_f32(Wd + (size_t)kk * Hdim + nt * 16, Hdim, lane);
      c0t = __builtin_amdgcn_wmma_f32_16x16x32_bf16(false, a0, false, b,
                                                    (short)0, c0t, false, false);
      c1t = __builtin_amdgcn_wmma_f32_16x16x32_bf16(false, a1, false, b,
                                                    (short)0, c1t, false, false);
    }
  }
  {
    int n = lane & 15, mb = (lane & 16) ? 8 : 0;
#pragma unroll
    for (int r = 0; r < 8; ++r) {
      part[wv * 512 +       (mb + r) * 16 + n] = c0t[r];
      part[wv * 512 + 256 + (mb + r) * 16 + n] = c1t[r];
    }
  }
  __syncthreads();
  for (int i = threadIdx.x; i < 512; i += 256) {
    float s = 0.0f;
#pragma unroll
    for (int w = 0; w < 8; ++w) s += part[w * 512 + i];
    int m = (i >> 8) * 16 + ((i >> 4) & 15);
    int n = nt * 16 + (i & 15);
    atomicAdd(&acc[m * Hdim + n], s);
  }
}

__global__ void finalize(const float* __restrict__ acc, const float* __restrict__ bd,
                         const float* __restrict__ h2, const float* __restrict__ c2,
                         float* __restrict__ out) {
  int i = blockIdx.x * 256 + threadIdx.x;
  if (i < BH) {
    float v = acc[i] + bd[i & (Hdim - 1)];
    out[i]          = v > 0.0f ? v : 0.0f;   // relu(flat@Wd + bd)
    out[BH + i]     = h2[i];
    out[2 * BH + i] = c2[i];
  }
}

// ------------------------------- launcher -----------------------------------

extern "C" void kernel_launch(void* const* d_in, const int* in_sizes, int n_in,
                              void* d_out, int out_size, void* d_ws, size_t ws_size,
                              hipStream_t stream) {
  (void)in_sizes; (void)n_in; (void)out_size; (void)ws_size;
  const float* X   = (const float*)d_in[0];   // [32][1068][256]
  const float* h0  = (const float*)d_in[1];   // [32][256]
  const float* c0  = (const float*)d_in[2];   // [32][256]
  const float* W1  = (const float*)d_in[3];   // [256][1024]
  const float* U1  = (const float*)d_in[4];
  const float* b1  = (const float*)d_in[5];
  const float* W2  = (const float*)d_in[6];
  const float* U2  = (const float*)d_in[7];
  const float* b2  = (const float*)d_in[8];
  const float* Wd  = (const float*)d_in[9];   // [273408][256]
  const float* bd  = (const float*)d_in[10];  // [256]
  float* out = (float*)d_out;

  char* ws = (char*)d_ws;
  size_t off = 0;
  auto alloc = [&](size_t bytes) -> void* {
    void* p = (void*)(ws + off);
    off += (bytes + 255) & ~(size_t)255;
    return p;
  };

  const size_t NX = (size_t)Bdim * T_STEPS * Hdim;   // 8,749,056
  const size_t NW = (size_t)Hdim * FH;               // 262,144

  __bf16* Xb    = (__bf16*)alloc(NX * 2);
  __bf16* W1p   = (__bf16*)alloc(NW * 2);
  __bf16* U1p   = (__bf16*)alloc(NW * 2);
  __bf16* W2p   = (__bf16*)alloc(NW * 2);
  __bf16* U2p   = (__bf16*)alloc(NW * 2);
  float*  XWp   = (float*)alloc((size_t)T_STEPS * Bdim * FH * 4);  // ~140 MB
  __bf16* seq1  = (__bf16*)alloc(NX * 2);
  __bf16* seq2  = (__bf16*)alloc(NX * 2);
  __bf16* hbuf1 = (__bf16*)alloc((size_t)2 * BH * 2);
  __bf16* hbuf2 = (__bf16*)alloc((size_t)2 * BH * 2);
  float*  h1f   = (float*)alloc(BH * 4);
  float*  c1f   = (float*)alloc(BH * 4);
  float*  h2f   = (float*)alloc(BH * 4);
  float*  c2f   = (float*)alloc(BH * 4);
  float*  acc   = (float*)alloc(BH * 4);
  int*    bars  = (int*)alloc(256);

  zero_init<<<32, 256, 0, stream>>>(acc, bars);

  cvt_f32_bf16<<<8192, 256, 0, stream>>>(X, Xb, (int)NX);
  pack_w<<<1024, 256, 0, stream>>>(W1, W1p);
  pack_w<<<1024, 256, 0, stream>>>(U1, U1p);
  pack_w<<<1024, 256, 0, stream>>>(W2, W2p);
  pack_w<<<1024, 256, 0, stream>>>(U2, U2p);

  const int gemmBlocks = MTILES * 64 / 8;   // 17088

  // layer 1
  gemm_xw<<<gemmBlocks, 256, 0, stream>>>(Xb, W1p, b1, XWp, 1);
  init_h<<<32, 256, 0, stream>>>(h0, hbuf1);
  lstm_rec<<<RBLK, 256, 0, stream>>>(XWp, U1p, c0, hbuf1, h1f, c1f, seq1,
                                     bars + 0, bars + 1);

  // layer 2: initial (h, c) = (h1, h1)  [faithful to reference quirk]
  gemm_xw<<<gemmBlocks, 256, 0, stream>>>(seq1, W2p, b2, XWp, 0);
  init_h<<<32, 256, 0, stream>>>(h1f, hbuf2);
  lstm_rec<<<RBLK, 256, 0, stream>>>(XWp, U2p, h1f, hbuf2, h2f, c2f, seq2,
                                     bars + 2, bars + 3);

  // dense + relu, emit (core_output, h2, c2)
  dense_partial<<<dim3(134, 16), 256, 0, stream>>>(seq2, Wd, acc);
  finalize<<<32, 256, 0, stream>>>(acc, bd, h2f, c2f, out);
}
